// CausalSelfAttention_42339787604444
// MI455X (gfx1250) — matrix operand order repaired and checked by reading, not compiled, not run on hardware
//
#include <hip/hip_runtime.h>
#include <hip/hip_bf16.h>
#include <stdint.h>

// ---------------- CDNA5 (gfx1250) wave32 WMMA types ----------------
typedef __attribute__((ext_vector_type(16))) __bf16 v16bf;
typedef __attribute__((ext_vector_type(8)))  __bf16 v8bf;
typedef __attribute__((ext_vector_type(8)))  float  v8f;

#define BATCH 4
#define SEQ   2048
#define CEMB  1024
#define C3    (3 * CEMB)
#define NHEAD 16
#define HDIM  64
#define NEG_BIG (-1.0e30f)

// fp32 -> bf16 round-to-nearest-even
static __device__ __forceinline__ __bf16 f2bf(float f) {
    uint32_t u = __builtin_bit_cast(uint32_t, f);
    uint32_t r = (u + 0x7FFFu + ((u >> 16) & 1u)) >> 16;
    uint16_t h = (uint16_t)r;
    return __builtin_bit_cast(__bf16, h);
}

// A-fragment (16x32 bf16): lane holds row m=lane&15; element i -> k =
// (lane>>4)*16 + i.  Caller passes p = rowbase + (lane>>4)*8; a further
// (lane>>4)*8 is added here, so the 16 elements are contiguous.
static __device__ __forceinline__ v16bf ldfragA(const __bf16* p) {
    p += ((threadIdx.x >> 4) & 1) * 8;
    v8bf lo = *reinterpret_cast<const v8bf*>(p);
    v8bf hi = *reinterpret_cast<const v8bf*>(p + 8);
    v16bf r;
#pragma unroll
    for (int i = 0; i < 8; ++i) { r[i] = lo[i]; r[i + 8] = hi[i]; }
    return r;
}

// B-fragment (32x16 bf16): lane holds col n=lane&15; element i -> k =
// (lane>>4)*16 + i.  Caller passes p = colbase + (lane>>4)*16 : 16 contiguous.
static __device__ __forceinline__ v16bf ldfragB(const __bf16* p) {
    v8bf lo = *reinterpret_cast<const v8bf*>(p);
    v8bf hi = *reinterpret_cast<const v8bf*>(p + 8);
    v16bf r;
#pragma unroll
    for (int i = 0; i < 8; ++i) { r[i] = lo[i]; r[i + 8] = hi[i]; }
    return r;
}

static __device__ __forceinline__ v8f wmma_bf16(v16bf a, v16bf b, v8f c) {
    // (neg_a, A, neg_b, B, c_mod, C, reuse_a, reuse_b)
    return __builtin_amdgcn_wmma_f32_16x16x32_bf16(false, a, false, b,
                                                   (short)0, c, false, false);
}

// ---------------- conversion kernels ----------------
__global__ void k_f32_to_bf16(const float* __restrict__ in,
                              __bf16* __restrict__ out, int n) {
    int i = blockIdx.x * blockDim.x + threadIdx.x;
    if (i < n) out[i] = f2bf(in[i]);
}

// in: [K][N] fp32 row-major  ->  out: [N][K] bf16
__global__ void k_transpose_bf16(const float* __restrict__ in,
                                 __bf16* __restrict__ out, int K, int N) {
    int i = blockIdx.x * blockDim.x + threadIdx.x;
    if (i < K * N) {
        int k = i / N, n = i % N;
        out[(size_t)n * K + k] = f2bf(in[i]);
    }
}

// ---------------- QKV GEMM: [8192x1024] @ [1024x3072] + bias ----------------
// 32x32 output tile per wave (2x2 WMMA register blocking).
// Scatters Q (prescaled by 1/8), K into [B,H,T,D] bf16; V into [B,H,D,T] bf16.
__global__ void k_gemm_qkv(const __bf16* __restrict__ xb,
                           const __bf16* __restrict__ wT,   // [3072][1024]
                           const float*  __restrict__ bias, // [3072]
                           __bf16* __restrict__ Qb, __bf16* __restrict__ Kb,
                           __bf16* __restrict__ Vt) {
    int lane = threadIdx.x & 31, wave = threadIdx.x >> 5;
    int ln = lane & 15, half = lane >> 4;
    int row0 = blockIdx.x * 32;                 // M tile (B*T rows)
    int n0   = (blockIdx.y * 8 + wave) * 32;    // N tile (3C cols)
    const __bf16* arow0 = xb + (size_t)(row0 + ln) * CEMB + half * 8;
    const __bf16* arow1 = xb + (size_t)(row0 + 16 + ln) * CEMB + half * 8;
    const __bf16* brow0 = wT + (size_t)(n0 + ln) * CEMB + half * 16;
    const __bf16* brow1 = wT + (size_t)(n0 + 16 + ln) * CEMB + half * 16;
    v8f acc[2][2] = {};
#pragma unroll 4
    for (int k0 = 0; k0 < CEMB; k0 += 32) {
        v16bf a0 = ldfragA(arow0 + k0);
        v16bf a1 = ldfragA(arow1 + k0);
        v16bf b0 = ldfragB(brow0 + k0);
        v16bf b1 = ldfragB(brow1 + k0);
        acc[0][0] = wmma_bf16(a0, b0, acc[0][0]);
        acc[0][1] = wmma_bf16(a0, b1, acc[0][1]);
        acc[1][0] = wmma_bf16(a1, b0, acc[1][0]);
        acc[1][1] = wmma_bf16(a1, b1, acc[1][1]);
    }
#pragma unroll
    for (int nt = 0; nt < 2; ++nt) {
        int ncol = n0 + nt * 16 + ln;
        float bv = bias[ncol];
        int sec = ncol >> 10;             // 0:Q 1:K 2:V
        int cc  = ncol & (CEMB - 1);
        int h = cc >> 6, d = cc & 63;
#pragma unroll
        for (int mt = 0; mt < 2; ++mt) {
#pragma unroll
            for (int v = 0; v < 8; ++v) {
                int row = row0 + mt * 16 + v + half * 8;
                int b_  = row >> 11, t = row & (SEQ - 1);
                size_t bh = (size_t)b_ * NHEAD + h;
                float val = acc[mt][nt][v] + bv;
                if (sec == 0)      Qb[(bh * SEQ + t) * HDIM + d] = f2bf(val * 0.125f);
                else if (sec == 1) Kb[(bh * SEQ + t) * HDIM + d] = f2bf(val);
                else               Vt[(bh * HDIM + d) * SEQ + t] = f2bf(val);
            }
        }
    }
}

// ---------------- flash attention: one wave per 32-query tile ----------------
// Two 16-row q-tiles per wave share each K/V fragment (halves K/V load traffic).
__global__ void k_attn(const __bf16* __restrict__ Qb,  // [B,H,T,D]
                       const __bf16* __restrict__ Kb,  // [B,H,T,D]
                       const __bf16* __restrict__ Vt,  // [B,H,D,T]
                       __bf16* __restrict__ Ya) {      // [B,T,C]
    __shared__ __align__(16) __bf16 Plds[8][2][16][32];
    int lane = threadIdx.x & 31, wave = threadIdx.x >> 5;
    int ln = lane & 15, half = lane >> 4;
    int bh = blockIdx.x;
    int b_ = bh >> 4, h = bh & 15;
    int q0 = (blockIdx.y * 8 + wave) * 32;
    const __bf16* Qp = Qb + (size_t)bh * SEQ * HDIM;
    const __bf16* Kp = Kb + (size_t)bh * SEQ * HDIM;
    const __bf16* Vp = Vt + (size_t)bh * HDIM * SEQ;

    v16bf qa[2][2];
#pragma unroll
    for (int qt = 0; qt < 2; ++qt) {
        qa[qt][0] = ldfragA(Qp + (size_t)(q0 + qt * 16 + ln) * HDIM + half * 8);
        qa[qt][1] = ldfragA(Qp + (size_t)(q0 + qt * 16 + ln) * HDIM + 32 + half * 8);
    }

    float mrow[2][8], lrow[2][8];
    v8f o[2][4] = {};
#pragma unroll
    for (int qt = 0; qt < 2; ++qt)
#pragma unroll
        for (int v = 0; v < 8; ++v) { mrow[qt][v] = NEG_BIG; lrow[qt][v] = 0.0f; }

    int nkb = (q0 + 31) / 32 + 1;   // causal: key blocks of 32 covering <= q0+31
    for (int kb = 0; kb < nkb; ++kb) {
        int k0 = kb * 32;
        if (kb + 1 < nkb) {
            // prefetch next key block (emits global_prefetch_b8)
            __builtin_prefetch(Kp + (size_t)(k0 + 32 + lane) * HDIM, 0, 2);
        }
        // K^T B-fragments: col n = key index, k-dim = d  (shared by both q-tiles)
        v16bf kf[2][2];
        kf[0][0] = ldfragB(Kp + (size_t)(k0 + ln) * HDIM + half * 16);
        kf[0][1] = ldfragB(Kp + (size_t)(k0 + ln) * HDIM + 32 + half * 16);
        kf[1][0] = ldfragB(Kp + (size_t)(k0 + 16 + ln) * HDIM + half * 16);
        kf[1][1] = ldfragB(Kp + (size_t)(k0 + 16 + ln) * HDIM + 32 + half * 16);

        int tk0 = k0 + ln, tk1 = k0 + 16 + ln;
#pragma unroll
        for (int qt = 0; qt < 2; ++qt) {
            v8f s0 = {}, s1 = {};
            s0 = wmma_bf16(qa[qt][0], kf[0][0], s0);
            s0 = wmma_bf16(qa[qt][1], kf[0][1], s0);   // keys k0..k0+15
            s1 = wmma_bf16(qa[qt][0], kf[1][0], s1);
            s1 = wmma_bf16(qa[qt][1], kf[1][1], s1);   // keys k0+16..k0+31
#pragma unroll
            for (int v = 0; v < 8; ++v) {
                int tq = q0 + qt * 16 + v + half * 8;
                float a = (tk0 <= tq) ? s0[v] : NEG_BIG;  // causal mask
                float c = (tk1 <= tq) ? s1[v] : NEG_BIG;
                float mx = fmaxf(a, c);
                mx = fmaxf(mx, __shfl_xor(mx, 1, 32));
                mx = fmaxf(mx, __shfl_xor(mx, 2, 32));
                mx = fmaxf(mx, __shfl_xor(mx, 4, 32));
                mx = fmaxf(mx, __shfl_xor(mx, 8, 32));
                float mnew  = fmaxf(mrow[qt][v], mx);
                float scale = __expf(mrow[qt][v] - mnew);
                float p0 = __expf(a - mnew);
                float p1 = __expf(c - mnew);
                float rs = p0 + p1;
                rs += __shfl_xor(rs, 1, 32);
                rs += __shfl_xor(rs, 2, 32);
                rs += __shfl_xor(rs, 4, 32);
                rs += __shfl_xor(rs, 8, 32);
                lrow[qt][v] = lrow[qt][v] * scale + rs;
                mrow[qt][v] = mnew;
                o[qt][0][v] *= scale; o[qt][1][v] *= scale;
                o[qt][2][v] *= scale; o[qt][3][v] *= scale;
                int m = v + half * 8;
                Plds[wave][qt][m][ln]      = f2bf(p0);
                Plds[wave][qt][m][16 + ln] = f2bf(p1);
            }
        }
        __asm__ volatile("s_wait_dscnt 0x0" ::: "memory");
        // V B-fragments from [d][t]: col n = d, k-dim = key (contiguous t),
        // shared by both q-tiles.
        v16bf vb[4];
#pragma unroll
        for (int dt = 0; dt < 4; ++dt)
            vb[dt] = ldfragB(Vp + (size_t)(dt * 16 + ln) * SEQ + k0 + half * 16);
#pragma unroll
        for (int qt = 0; qt < 2; ++qt) {
            v16bf pa = ldfragA(&Plds[wave][qt][ln][half * 8]);
            o[qt][0] = wmma_bf16(pa, vb[0], o[qt][0]);
            o[qt][1] = wmma_bf16(pa, vb[1], o[qt][1]);
            o[qt][2] = wmma_bf16(pa, vb[2], o[qt][2]);
            o[qt][3] = wmma_bf16(pa, vb[3], o[qt][3]);
        }
    }
#pragma unroll
    for (int qt = 0; qt < 2; ++qt) {
#pragma unroll
        for (int v = 0; v < 8; ++v) {
            float inv = 1.0f / lrow[qt][v];
            int t = q0 + qt * 16 + v + half * 8;
            size_t base = ((size_t)b_ * SEQ + t) * CEMB + h * HDIM;
            Ya[base + 0 * 16 + ln] = f2bf(o[qt][0][v] * inv);
            Ya[base + 1 * 16 + ln] = f2bf(o[qt][1][v] * inv);
            Ya[base + 2 * 16 + ln] = f2bf(o[qt][2][v] * inv);
            Ya[base + 3 * 16 + ln] = f2bf(o[qt][3][v] * inv);
        }
    }
}

// ---------------- output projection: [8192x1024] @ [1024x1024] + bias ----------
// 32x32 output tile per wave (2x2 WMMA register blocking).
__global__ void k_gemm_proj(const __bf16* __restrict__ yb,   // [B*T][C]
                            const __bf16* __restrict__ wT,   // [C][C] (N-major)
                            const float*  __restrict__ bias, // [C]
                            float* __restrict__ out) {       // [B*T][C] fp32
    int lane = threadIdx.x & 31, wave = threadIdx.x >> 5;
    int ln = lane & 15, half = lane >> 4;
    int row0 = blockIdx.x * 32;
    int n0   = (blockIdx.y * 8 + wave) * 32;
    const __bf16* arow0 = yb + (size_t)(row0 + ln) * CEMB + half * 8;
    const __bf16* arow1 = yb + (size_t)(row0 + 16 + ln) * CEMB + half * 8;
    const __bf16* brow0 = wT + (size_t)(n0 + ln) * CEMB + half * 16;
    const __bf16* brow1 = wT + (size_t)(n0 + 16 + ln) * CEMB + half * 16;
    v8f acc[2][2] = {};
#pragma unroll 4
    for (int k0 = 0; k0 < CEMB; k0 += 32) {
        v16bf a0 = ldfragA(arow0 + k0);
        v16bf a1 = ldfragA(arow1 + k0);
        v16bf b0 = ldfragB(brow0 + k0);
        v16bf b1 = ldfragB(brow1 + k0);
        acc[0][0] = wmma_bf16(a0, b0, acc[0][0]);
        acc[0][1] = wmma_bf16(a0, b1, acc[0][1]);
        acc[1][0] = wmma_bf16(a1, b0, acc[1][0]);
        acc[1][1] = wmma_bf16(a1, b1, acc[1][1]);
    }
#pragma unroll
    for (int nt = 0; nt < 2; ++nt) {
        int ncol = n0 + nt * 16 + ln;
        float bv = bias[ncol];
#pragma unroll
        for (int mt = 0; mt < 2; ++mt) {
#pragma unroll
            for (int v = 0; v < 8; ++v) {
                int row = row0 + mt * 16 + v + half * 8;
                out[(size_t)row * CEMB + ncol] = acc[mt][nt][v] + bv;
            }
        }
    }
}

// ---------------- host-side launch ----------------
extern "C" void kernel_launch(void* const* d_in, const int* in_sizes, int n_in,
                              void* d_out, int out_size, void* d_ws, size_t ws_size,
                              hipStream_t stream) {
    const float* x      = (const float*)d_in[0];
    const float* w_attn = (const float*)d_in[1];
    const float* b_attn = (const float*)d_in[2];
    const float* w_proj = (const float*)d_in[3];
    const float* b_proj = (const float*)d_in[4];
    float* out = (float*)d_out;

    const size_t XB = (size_t)BATCH * SEQ * CEMB;       // 8,388,608 elements
    char* ws = (char*)d_ws;
    __bf16* xb  = (__bf16*)ws;  ws += XB * 2;
    __bf16* waT = (__bf16*)ws;  ws += (size_t)C3 * CEMB * 2;
    __bf16* wpT = (__bf16*)ws;  ws += (size_t)CEMB * CEMB * 2;
    __bf16* Qb  = (__bf16*)ws;  ws += XB * 2;           // [B,H,T,D]
    __bf16* Kb  = (__bf16*)ws;  ws += XB * 2;           // [B,H,T,D]
    __bf16* Vt  = (__bf16*)ws;  ws += XB * 2;           // [B,H,D,T]
    __bf16* Ya  = (__bf16*)ws;  ws += XB * 2;           // [B,T,C]

    k_f32_to_bf16<<<(unsigned)((XB + 255) / 256), 256, 0, stream>>>(x, xb, (int)XB);
    k_transpose_bf16<<<(unsigned)(((size_t)CEMB * C3 + 255) / 256), 256, 0, stream>>>(
        w_attn, waT, CEMB, C3);
    k_transpose_bf16<<<(unsigned)(((size_t)CEMB * CEMB + 255) / 256), 256, 0, stream>>>(
        w_proj, wpT, CEMB, CEMB);

    dim3 gq(BATCH * SEQ / 32, C3 / 256);     // (256, 12), 8 waves/block
    k_gemm_qkv<<<gq, 256, 0, stream>>>(xb, waT, b_attn, Qb, Kb, Vt);

    dim3 ga(BATCH * NHEAD, SEQ / 256);       // (64, 8), 8 x 32-query tiles/block
    k_attn<<<ga, 256, 0, stream>>>(Qb, Kb, Vt, Ya);

    dim3 gp(BATCH * SEQ / 32, CEMB / 256);   // (256, 4)
    k_gemm_proj<<<gp, 256, 0, stream>>>(Ya, wpT, b_proj, out);
}
